// GRUNet_41472204210285
// MI455X (gfx1250) — compile-verified, matrix operand order checked by reading, not run on hardware
//
#include <hip/hip_runtime.h>
#include <hip/hip_bf16.h>

#define LN_EPS 1e-5f

typedef __attribute__((ext_vector_type(16))) __bf16 bf16x16;
typedef __attribute__((ext_vector_type(8)))  float  f32x8;

union TileU { bf16x16 v; unsigned int u[8]; };

__device__ __forceinline__ unsigned short f2bf(float f) {
  unsigned int u = __float_as_uint(f);
  u += 0x7FFFu + ((u >> 16) & 1u);          // round-to-nearest-even
  return (unsigned short)(u >> 16);
}
__device__ __forceinline__ float bf2f(unsigned short h) {
  return __uint_as_float(((unsigned int)h) << 16);
}
__device__ __forceinline__ float sigmoidf(float x) { return 1.0f / (1.0f + __expf(-x)); }

// Load one 16x32 bf16 fragment (A or B^T) from a row-major [rows x ldk] matrix.
// ISA 16-bit A layout: lanes 0-15 hold M=0-15 / K-base 0, lanes 16-31 K-base 8;
// dword i holds K pair ((i>>2)<<4) + kb + ((i&3)<<1).
__device__ __forceinline__ bf16x16 load_tile(const unsigned short* base, int row0,
                                             int ldk, int k0, int lane) {
  const int mr = lane & 15;
  const int kb = (lane & 16) ? 8 : 0;
  const unsigned short* p = base + (size_t)(row0 + mr) * (size_t)ldk + (size_t)(k0 + kb);
  TileU t;
#pragma unroll
  for (int i = 0; i < 8; ++i) {
    const int k = ((i >> 2) << 4) + ((i & 3) << 1);
    t.u[i] = *reinterpret_cast<const unsigned int*>(p + k);
  }
  return t.v;
}

// ---------------- generic WMMA GEMM: C(MxN) = A(MxK) * B(NxK)^T -------------
// One wave per 16x16 output tile. Stores bf16 (Cbf) or f32+bias (Cf).
__global__ void wmma_gemm_bf16(const unsigned short* __restrict__ A,
                               const unsigned short* __restrict__ B,
                               unsigned short* __restrict__ Cbf,
                               float* __restrict__ Cf,
                               const float* __restrict__ bias,
                               int M, int N, int K) {
  const int lane = threadIdx.x & 31;
  const int wave = blockIdx.x * (blockDim.x >> 5) + (threadIdx.x >> 5);
  const int ntn  = N >> 4;
  const int total = (M >> 4) * ntn;
  if (wave >= total) return;                 // wave-uniform: EXEC stays all-ones
  const int tm = wave / ntn;
  const int tn = wave - tm * ntn;
  f32x8 acc = {};
  for (int k0 = 0; k0 < K; k0 += 32) {
    bf16x16 a = load_tile(A, tm << 4, K, k0, lane);
    bf16x16 b = load_tile(B, tn << 4, K, k0, lane);
    acc = __builtin_amdgcn_wmma_f32_16x16x32_bf16(false, a, false, b,
                                                  (short)0, acc, false, false);
  }
  const int nc = lane & 15;
  const int rb = (lane & 16) ? 8 : 0;
  const int col = (tn << 4) + nc;
  if (Cbf) {
#pragma unroll
    for (int r = 0; r < 8; ++r)
      Cbf[(size_t)((tm << 4) + rb + r) * (size_t)N + col] = f2bf(acc[r]);
  } else {
    const float bv = bias ? bias[col] : 0.0f;
#pragma unroll
    for (int r = 0; r < 8; ++r)
      Cf[(size_t)((tm << 4) + rb + r) * (size_t)N + col] = acc[r] + bv;
  }
}

// ---------------- grid-wide barrier (persistent kernel) ---------------------
__device__ __forceinline__ void grid_barrier(unsigned int* cnt, unsigned int target) {
  __threadfence();                            // make prior stores agent-visible
  __syncthreads();
  if (threadIdx.x == 0) {
    __hip_atomic_fetch_add(cnt, 1u, __ATOMIC_RELEASE, __HIP_MEMORY_SCOPE_AGENT);
    while (__hip_atomic_load(cnt, __ATOMIC_ACQUIRE, __HIP_MEMORY_SCOPE_AGENT) < target)
      __builtin_amdgcn_s_sleep(2);
  }
  __syncthreads();
}

// ---------------- persistent GRU layer --------------------------------------
// gi: (S*B, 768) bf16 (no biases), whh: (768,256) bf16, h: (208,256) bf16 (padded),
// gh: (208,768) f32 scratch, hs: (S*B,256) bf16 output.
__global__ void gru_layer_persistent(const unsigned short* __restrict__ gi,
                                     const unsigned short* __restrict__ whh,
                                     const float* __restrict__ b_ih,
                                     const float* __restrict__ b_hh,
                                     unsigned short* __restrict__ h,
                                     float* __restrict__ gh,
                                     unsigned short* __restrict__ hs,
                                     unsigned int* __restrict__ cnt,
                                     int nblocks) {
  const int S = 512, Bt = 200, H = 256, G = 768, MP = 208;
  const int lane  = threadIdx.x & 31;
  const int gwave = blockIdx.x * (blockDim.x >> 5) + (threadIdx.x >> 5);
  const int tid   = blockIdx.x * blockDim.x + threadIdx.x;
  const int nthr  = nblocks * blockDim.x;
  unsigned int gen = 0;

  // h0 = 0 (including the 8 pad rows so GEMM tiles read zeros)
  for (int i = tid; i < MP * H; i += nthr) h[i] = 0;
  gen += nblocks; grid_barrier(cnt, gen);

  for (int t = 0; t < S; ++t) {
    // ---- gh = h @ whh^T : 13 x 48 tiles of 16x16, K=256 (8 WMMAs each) ----
    if (gwave < 13 * 48) {
      const int tm = gwave / 48;
      const int tn = gwave - tm * 48;
      f32x8 acc = {};
#pragma unroll
      for (int k0 = 0; k0 < H; k0 += 32) {
        bf16x16 a = load_tile(h,   tm << 4, H, k0, lane);
        bf16x16 b = load_tile(whh, tn << 4, H, k0, lane);
        acc = __builtin_amdgcn_wmma_f32_16x16x32_bf16(false, a, false, b,
                                                      (short)0, acc, false, false);
      }
      const int nc = lane & 15;
      const int rb = (lane & 16) ? 8 : 0;
#pragma unroll
      for (int r = 0; r < 8; ++r)
        gh[((tm << 4) + rb + r) * G + (tn << 4) + nc] = acc[r];
    }
    gen += nblocks; grid_barrier(cnt, gen);

    // ---- gates + state update (torch GRU order [r,z,n]) ----
    const size_t gib0 = (size_t)t * Bt * G;
    for (int e = tid; e < Bt * H; e += nthr) {
      const int m = e >> 8;                  // H == 256
      const int j = e & 255;
      const size_t gib = gib0 + (size_t)m * G;
      const float ir = bf2f(gi[gib + j])         + b_ih[j];
      const float iz = bf2f(gi[gib + H + j])     + b_ih[H + j];
      const float in = bf2f(gi[gib + 2 * H + j]) + b_ih[2 * H + j];
      const float hr = gh[m * G + j]         + b_hh[j];
      const float hz = gh[m * G + H + j]     + b_hh[H + j];
      const float hn = gh[m * G + 2 * H + j] + b_hh[2 * H + j];
      const float r = sigmoidf(ir + hr);
      const float z = sigmoidf(iz + hz);
      const float n = tanhf(in + r * hn);
      const float hv = (1.0f - z) * n + z * bf2f(h[e]);
      const unsigned short hb = f2bf(hv);
      h[e] = hb;
      hs[(size_t)t * Bt * H + e] = hb;
    }
    gen += nblocks; grid_barrier(cnt, gen);
  }
}

// ---------------- LayerNorm(H) + ReLU ---------------------------------------
__global__ void ln_relu(const unsigned short* __restrict__ hsv,
                        const float* __restrict__ w, const float* __restrict__ b,
                        unsigned short* __restrict__ outp) {
  __shared__ float red[256];
  const int row = blockIdx.x;
  const int j = threadIdx.x;
  const float v = bf2f(hsv[(size_t)row * 256 + j]);
  red[j] = v; __syncthreads();
  for (int s = 128; s > 0; s >>= 1) { if (j < s) red[j] += red[j + s]; __syncthreads(); }
  const float mu = red[0] * (1.0f / 256.0f);
  __syncthreads();
  const float d = v - mu;
  red[j] = d * d; __syncthreads();
  for (int s = 128; s > 0; s >>= 1) { if (j < s) red[j] += red[j + s]; __syncthreads(); }
  const float var = red[0] * (1.0f / 256.0f);
  float y = d * rsqrtf(var + LN_EPS) * w[j] + b[j];
  outp[(size_t)row * 256 + j] = f2bf(fmaxf(y, 0.0f));
}

// ---------------- BatchNorm stats over axis 0 -------------------------------
__global__ void bn_stats(const float* __restrict__ y, float* __restrict__ stats) {
  const int c = blockIdx.x * blockDim.x + threadIdx.x;
  if (c >= 512) return;
  float s = 0.0f, sq = 0.0f;
  for (int r = 0; r < 512; ++r) { const float v = y[r * 512 + c]; s += v; sq += v * v; }
  const float mu = s * (1.0f / 512.0f);
  stats[c] = mu;
  stats[512 + c] = sq * (1.0f / 512.0f) - mu * mu;
}

// ---------------- BN apply + ReLU + FC3 -------------------------------------
__global__ void bn_fc3(const float* __restrict__ y, const float* __restrict__ stats,
                       const float* __restrict__ g, const float* __restrict__ b,
                       const float* __restrict__ w3, const float* __restrict__ b3,
                       float* __restrict__ out) {
  const int idx = blockIdx.x * blockDim.x + threadIdx.x;
  if (idx >= 512 * 3) return;
  const int r = idx / 3, o = idx - r * 3;
  float acc = b3[o];
  for (int c = 0; c < 512; ++c) {
    float v = (y[r * 512 + c] - stats[c]) * rsqrtf(stats[512 + c] + LN_EPS) * g[c] + b[c];
    acc += fmaxf(v, 0.0f) * w3[o * 512 + c];
  }
  out[idx] = acc;
}

// ---------------- helpers ---------------------------------------------------
__global__ void cvt_f32_bf16(const float* __restrict__ in,
                             unsigned short* __restrict__ out, int n) {
  const int i = blockIdx.x * blockDim.x + threadIdx.x;
  if (i < n) out[i] = f2bf(in[i]);
}
__global__ void zero_u32(unsigned int* p, int n) {
  const int i = blockIdx.x * blockDim.x + threadIdx.x;
  if (i < n) p[i] = 0;
}

extern "C" void kernel_launch(void* const* d_in, const int* in_sizes, int n_in,
                              void* d_out, int out_size, void* d_ws, size_t ws_size,
                              hipStream_t stream) {
  (void)in_sizes; (void)n_in; (void)out_size; (void)ws_size;
  const float* x     = (const float*)d_in[0];
  const float* w_ih0 = (const float*)d_in[1];
  const float* w_hh0 = (const float*)d_in[2];
  const float* b_ih0 = (const float*)d_in[3];
  const float* b_hh0 = (const float*)d_in[4];
  const float* w_ih1 = (const float*)d_in[5];
  const float* w_hh1 = (const float*)d_in[6];
  const float* b_ih1 = (const float*)d_in[7];
  const float* b_hh1 = (const float*)d_in[8];
  const float* ln_w  = (const float*)d_in[9];
  const float* ln_b  = (const float*)d_in[10];
  const float* fc2_w = (const float*)d_in[11];
  const float* fc2_b = (const float*)d_in[12];
  const float* bn_g  = (const float*)d_in[13];
  const float* bn_b  = (const float*)d_in[14];
  const float* fc3_w = (const float*)d_in[15];
  const float* fc3_b = (const float*)d_in[16];
  float* out = (float*)d_out;

  const int S = 512, Bt = 200, D = 64, H = 256, G = 768;
  const int SB = S * Bt, F = 512, KF = Bt * H;

  char* wsp = (char*)d_ws;
  size_t off = 0;
  auto alloc = [&](size_t bytes) -> void* {
    void* p = wsp + off;
    off = (off + bytes + 255) & ~(size_t)255;
    return p;
  };
  unsigned int*   cnt     = (unsigned int*)  alloc(64 * 4);
  float*          stats   = (float*)         alloc((size_t)2 * F * 4);
  unsigned short* x_bf    = (unsigned short*)alloc((size_t)SB * D * 2);
  unsigned short* wih0_bf = (unsigned short*)alloc((size_t)G * D * 2);
  unsigned short* whh0_bf = (unsigned short*)alloc((size_t)G * H * 2);
  unsigned short* wih1_bf = (unsigned short*)alloc((size_t)G * H * 2);
  unsigned short* whh1_bf = (unsigned short*)alloc((size_t)G * H * 2);
  unsigned short* fc2w_bf = (unsigned short*)alloc((size_t)F * KF * 2);
  unsigned short* gi_bf   = (unsigned short*)alloc((size_t)SB * G * 2);   // reused both layers
  unsigned short* h_bf    = (unsigned short*)alloc((size_t)208 * H * 2);  // padded 200->208
  float*          gh      = (float*)         alloc((size_t)208 * G * 4);
  unsigned short* hs0_bf  = (unsigned short*)alloc((size_t)SB * H * 2);
  unsigned short* hs1_bf  = (unsigned short*)alloc((size_t)SB * H * 2);
  unsigned short* hflat   = (unsigned short*)alloc((size_t)SB * H * 2);   // == (512, 51200)
  float*          y       = (float*)         alloc((size_t)F * F * 4);

  zero_u32<<<1, 64, 0, stream>>>(cnt, 64);

  auto cvt = [&](const float* src, unsigned short* dst, int n) {
    cvt_f32_bf16<<<(n + 255) / 256, 256, 0, stream>>>(src, dst, n);
  };
  cvt(x,     x_bf,    SB * D);
  cvt(w_ih0, wih0_bf, G * D);
  cvt(w_hh0, whh0_bf, G * H);
  cvt(w_ih1, wih1_bf, G * H);
  cvt(w_hh1, whh1_bf, G * H);
  cvt(fc2_w, fc2w_bf, F * KF);

  const int GRU_BLOCKS = 104;  // 104 blocks x 8 waves = 832 waves >= 624 tiles

  // gi0 = x @ w_ih0^T  (no bias; biases folded into gate phase)
  {
    const int tiles = (SB / 16) * (G / 16);
    wmma_gemm_bf16<<<(tiles + 7) / 8, 256, 0, stream>>>(
        x_bf, wih0_bf, gi_bf, nullptr, nullptr, SB, G, D);
  }
  gru_layer_persistent<<<GRU_BLOCKS, 256, 0, stream>>>(
      gi_bf, whh0_bf, b_ih0, b_hh0, h_bf, gh, hs0_bf, cnt + 0, GRU_BLOCKS);

  // gi1 = hs0 @ w_ih1^T
  {
    const int tiles = (SB / 16) * (G / 16);
    wmma_gemm_bf16<<<(tiles + 7) / 8, 256, 0, stream>>>(
        hs0_bf, wih1_bf, gi_bf, nullptr, nullptr, SB, G, H);
  }
  gru_layer_persistent<<<GRU_BLOCKS, 256, 0, stream>>>(
      gi_bf, whh1_bf, b_ih1, b_hh1, h_bf, gh, hs1_bf, cnt + 1, GRU_BLOCKS);

  // LayerNorm + ReLU -> hflat (same memory layout as reshape(S, B*H))
  ln_relu<<<SB, 256, 0, stream>>>(hs1_bf, ln_w, ln_b, hflat);

  // y = hflat(512 x 51200) @ fc2_w^T + fc2_b
  {
    const int tiles = (F / 16) * (F / 16);
    wmma_gemm_bf16<<<(tiles + 7) / 8, 256, 0, stream>>>(
        hflat, fc2w_bf, nullptr, y, fc2_b, F, F, KF);
  }
  bn_stats<<<2, 256, 0, stream>>>(y, stats);
  bn_fc3<<<6, 256, 0, stream>>>(y, stats, bn_g, bn_b, fc3_w, fc3_b, out);
}
